// SpatialSelfAttention_4982162063897
// MI455X (gfx1250) — matrix operand order repaired and checked by reading, not compiled, not run on hardware
//
#include <hip/hip_runtime.h>
#include <hip/hip_bf16.h>
#include <math.h>

// ---------------------------------------------------------------------------
// Types for CDNA5 WMMA (wave32): 16x16x32 bf16 -> f32
// ---------------------------------------------------------------------------
typedef __attribute__((ext_vector_type(16))) __bf16 bf16x16;
typedef __attribute__((ext_vector_type(8)))  float  f32x8;

__device__ inline f32x8 wmma_bf16(bf16x16 a, bf16x16 b, f32x8 c) {
    // (neg_a, A, neg_b, B, c_mod, C, reuse_a, reuse_b)
    return __builtin_amdgcn_wmma_f32_16x16x32_bf16(false, a, false, b,
                                                   (short)0, c, false, false);
}

__device__ inline f32x8 zero8() {
    f32x8 v = {0.f, 0.f, 0.f, 0.f, 0.f, 0.f, 0.f, 0.f};
    return v;
}

// Load 16 bf16 (two aligned 16B chunks) into a WMMA operand register block.
__device__ inline bf16x16 ld_bf16x16(const __bf16* p0, const __bf16* p1) {
    union { uint4 u[2]; bf16x16 v; } t;
    t.u[0] = *(const uint4*)p0;
    t.u[1] = *(const uint4*)p1;
    return t.v;
}

// A operand (16x32, row-major source): lane -> row = lane%16,
// element e -> k = koff + (e<8 ? e : 16+e-8), koff = (lane&16)?8:0.
__device__ inline bf16x16 load_A(const __bf16* row_d0, int lane) {
    const int koff = (lane & 16) ? 8 : 0;
    return ld_bf16x16(row_d0 + koff, row_d0 + koff + 16);
}

// B operand (32x16): lane -> n = lane%16, element e -> k = kb + (e<8 ? e : 16+e-8),
// kb = (lane&16)?8:0.  Source must have K contiguous for fixed n.
__device__ inline bf16x16 load_B(const __bf16* row_d0, int lane) {
    const int kb = (lane & 16) ? 8 : 0;
    return ld_bf16x16(row_d0 + kb, row_d0 + kb + 16);
}

// A operand built from f32 source with on-the-fly bf16 conversion.
__device__ inline bf16x16 load_A_f32(const float* row_d0, int lane) {
    const int koff = (lane & 16) ? 8 : 0;
    const float* p0 = row_d0 + koff;
    const float* p1 = row_d0 + koff + 16;
    bf16x16 v;
#pragma unroll
    for (int i = 0; i < 8; ++i) v[i] = (__bf16)p0[i];
#pragma unroll
    for (int i = 0; i < 8; ++i) v[8 + i] = (__bf16)p1[i];
    return v;
}

#define N_TOK 8192
#define D_DIM 512

// ---------------------------------------------------------------------------
// Kernel 1: tiled transpose f32 -> bf16   out[c][r] = (bf16) in[r][c]
// grid = (C/32, R/32), block = 256
// ---------------------------------------------------------------------------
__global__ __launch_bounds__(256)
void transpose_to_bf16(const float* __restrict__ in, __bf16* __restrict__ out,
                       int R, int C) {
    __shared__ float tile[32][33];
    const int bx = blockIdx.x * 32;   // input col base
    const int by = blockIdx.y * 32;   // input row base
    const int x  = threadIdx.x & 31;
    const int y  = threadIdx.x >> 5;  // 0..7
#pragma unroll
    for (int i = 0; i < 4; ++i) {
        int r = y + i * 8;
        tile[r][x] = in[(size_t)(by + r) * C + bx + x];
    }
    __syncthreads();
#pragma unroll
    for (int i = 0; i < 4; ++i) {
        int r = y + i * 8;  // output row within tile (= input col)
        out[(size_t)(bx + r) * R + by + x] = (__bf16)tile[x][r];
    }
}

// ---------------------------------------------------------------------------
// Kernel 2: Qb = (H @ Wq + bq) * (1/sqrt(512)) ; Kb = H @ Wk + bk   (bf16 out)
// One wave computes a 16x64 output tile.  8192 waves total -> grid 1024 x 256.
// ---------------------------------------------------------------------------
__global__ __launch_bounds__(256)
void qk_gemm(const float* __restrict__ H,
             const __bf16* __restrict__ WqT, const float* __restrict__ bq,
             const __bf16* __restrict__ WkT, const float* __restrict__ bk,
             __bf16* __restrict__ Qb, __bf16* __restrict__ Kb) {
    const int lane = threadIdx.x & 31;
    int wid = (blockIdx.x * blockDim.x + threadIdx.x) >> 5;
    const int matrix = wid / 4096;           // 0 = Q, 1 = K
    wid %= 4096;
    const int mt  = wid >> 3;                // row tile (16 rows)
    const int nt4 = wid & 7;                 // 64-col group

    const __bf16* WT   = matrix ? WkT : WqT;
    const float*  bias = matrix ? bk  : bq;
    __bf16*       Out  = matrix ? Kb  : Qb;
    const float   scale = matrix ? 1.0f : 0.04419417382415922f; // 1/sqrt(512)

    const int lm  = lane & 15;
    const int lhi = (lane >> 4) & 1;

    f32x8 acc[4];
#pragma unroll
    for (int j = 0; j < 4; ++j) acc[j] = zero8();

    const float* hrow = H + (size_t)(mt * 16 + lm) * D_DIM;
#pragma unroll 4
    for (int d0 = 0; d0 < D_DIM; d0 += 32) {
        bf16x16 a = load_A_f32(hrow + d0, lane);
#pragma unroll
        for (int j = 0; j < 4; ++j) {
            int n = nt4 * 64 + j * 16 + lm;
            bf16x16 b = load_B(WT + (size_t)n * D_DIM + d0, lane);
            acc[j] = wmma_bf16(a, b, acc[j]);
        }
    }

#pragma unroll
    for (int j = 0; j < 4; ++j) {
        int n = nt4 * 64 + j * 16 + lm;
        float bv = bias[n];
#pragma unroll
        for (int e = 0; e < 8; ++e) {
            int row = mt * 16 + e + 8 * lhi;
            Out[(size_t)row * D_DIM + n] = (__bf16)((acc[j][e] + bv) * scale);
        }
    }
}

// ---------------------------------------------------------------------------
// Kernel 3: fused flash attention + residual
//   out = softmax(zero_diag(Qs @ K^T)) @ H + H     (Qs already scaled)
// 256 workgroups (32 rows each) x 8 waves; column blocks of 128.
//   waves arranged 2 (row halves of 16) x 4 (score-col / d-col quarters).
// ---------------------------------------------------------------------------
#define SROW 132           // padded f32 stride for score tile
#define PROW 136           // padded bf16 stride for P tile

__global__ __launch_bounds__(256)
void flash_attn(const float* __restrict__ H,
                const __bf16* __restrict__ Qb,
                const __bf16* __restrict__ Kb,
                const __bf16* __restrict__ HbT,
                float* __restrict__ out) {
    __shared__ float  Sld[32 * SROW];
    __shared__ __bf16 Pld[32 * PROW];
    __shared__ float  mrow[32], lrow[32], arow[32];

    const int tid  = threadIdx.x;
    const int lane = tid & 31;
    const int w    = tid >> 5;          // wave 0..7
    const int rsub = w >> 2;            // 0..1  (16-row half)
    const int csub = w & 3;             // 0..3  (quarter)
    const int rb   = blockIdx.x * 32;   // global row base
    const int lm   = lane & 15;
    const int lhi  = (lane >> 4) & 1;
    const int rloc = rsub * 16 + 8 * lhi;  // tile-row base for C-layout elems

    if (tid < 32) { mrow[tid] = -INFINITY; lrow[tid] = 0.0f; }
    __syncthreads();

    f32x8 o[8];
#pragma unroll
    for (int j = 0; j < 8; ++j) o[j] = zero8();

    const __bf16* qrow = Qb + (size_t)(rb + rsub * 16 + lm) * D_DIM;

    for (int cb = 0; cb < N_TOK; cb += 128) {
        // ---- Phase A: score sub-tile 16x32 via WMMA ----
        f32x8 s0 = zero8(), s1 = zero8();
        const __bf16* krow0 = Kb + (size_t)(cb + csub * 32 + lm) * D_DIM;
        const __bf16* krow1 = krow0 + (size_t)16 * D_DIM;
#pragma unroll 4
        for (int d0 = 0; d0 < D_DIM; d0 += 32) {
            bf16x16 a  = load_A(qrow + d0, lane);
            bf16x16 b0 = load_B(krow0 + d0, lane);
            bf16x16 b1 = load_B(krow1 + d0, lane);
            s0 = wmma_bf16(a, b0, s0);
            s1 = wmma_bf16(a, b1, s1);
        }
        {
            const int c0 = csub * 32 + lm;
#pragma unroll
            for (int e = 0; e < 8; ++e) {
                Sld[(rloc + e) * SROW + c0]      = s0[e];
                Sld[(rloc + e) * SROW + c0 + 16] = s1[e];
            }
        }
        __syncthreads();

        // ---- Phase B: online softmax (8 lanes per row) ----
        {
            const int r    = tid >> 3;      // 0..31
            const int sub  = tid & 7;       // 16-col chunk
            const int grow = rb + r;
            float sv[16];
            float mx = -INFINITY;
#pragma unroll
            for (int i = 0; i < 16; ++i) {
                int c = sub * 16 + i;
                float sx = Sld[r * SROW + c];
                if (cb + c == grow) sx = 0.0f;   // zeroed diagonal
                sv[i] = sx;
                mx = fmaxf(mx, sx);
            }
            mx = fmaxf(mx, __shfl_xor(mx, 1));
            mx = fmaxf(mx, __shfl_xor(mx, 2));
            mx = fmaxf(mx, __shfl_xor(mx, 4));
            float mold = mrow[r];
            float mnew = fmaxf(mold, mx);
            float sum = 0.0f;
#pragma unroll
            for (int i = 0; i < 16; ++i) {
                float p = __expf(sv[i] - mnew);
                sum += p;
                Pld[r * PROW + sub * 16 + i] = (__bf16)p;
            }
            sum += __shfl_xor(sum, 1);
            sum += __shfl_xor(sum, 2);
            sum += __shfl_xor(sum, 4);
            if (sub == 0) {
                float al = __expf(mold - mnew);
                arow[r] = al;
                lrow[r] = lrow[r] * al + sum;
                mrow[r] = mnew;
            }
        }
        __syncthreads();

        // ---- Phase C: O = O*alpha + P @ H  (wave owns 16 rows x 128 d-cols)
        {
            float av[8];
#pragma unroll
            for (int e = 0; e < 8; ++e) av[e] = arow[rloc + e];
#pragma unroll
            for (int j = 0; j < 8; ++j)
#pragma unroll
                for (int e = 0; e < 8; ++e) o[j][e] *= av[e];

#pragma unroll
            for (int kb = 0; kb < 128; kb += 32) {
                bf16x16 a = load_A(Pld + (rsub * 16 + lm) * PROW + kb, lane);
#pragma unroll
                for (int j = 0; j < 8; ++j) {
                    int n = csub * 128 + j * 16 + lm;
                    bf16x16 b = load_B(HbT + (size_t)n * N_TOK + cb + kb, lane);
                    o[j] = wmma_bf16(a, b, o[j]);
                }
            }
        }
        __syncthreads();   // protect Sld/Pld/arow for next block
    }

    // ---- Epilogue: O/l + residual ----
    {
        float rinv[8];
#pragma unroll
        for (int e = 0; e < 8; ++e) rinv[e] = 1.0f / lrow[rloc + e];
#pragma unroll
        for (int j = 0; j < 8; ++j) {
            int n = csub * 128 + j * 16 + lm;
#pragma unroll
            for (int e = 0; e < 8; ++e) {
                int grow = rb + rloc + e;
                size_t idx = (size_t)grow * D_DIM + n;
                out[idx] = o[j][e] * rinv[e] + H[idx];
            }
        }
    }
}

// ---------------------------------------------------------------------------
// Host launcher.  Workspace layout (bf16):
//   [0, 8MB)      HbT  : 512 x 8192
//   [8MB, 8.5MB)  WqT  : 512 x 512
//   [8.5MB, 9MB)  WkT  : 512 x 512
//   [9MB, 17MB)   Qb   : 8192 x 512 (pre-scaled by 1/sqrt(512))
//   [17MB, 25MB)  Kb   : 8192 x 512
// ---------------------------------------------------------------------------
extern "C" void kernel_launch(void* const* d_in, const int* in_sizes, int n_in,
                              void* d_out, int out_size, void* d_ws, size_t ws_size,
                              hipStream_t stream) {
    (void)in_sizes; (void)n_in; (void)out_size; (void)ws_size;
    const float* H  = (const float*)d_in[0];
    const float* Wq = (const float*)d_in[1];
    const float* bq = (const float*)d_in[2];
    const float* Wk = (const float*)d_in[3];
    const float* bk = (const float*)d_in[4];
    float* out = (float*)d_out;

    char* ws = (char*)d_ws;
    __bf16* HbT = (__bf16*)(ws);
    __bf16* WqT = (__bf16*)(ws + (size_t)8 * 1024 * 1024);
    __bf16* WkT = WqT + (size_t)D_DIM * D_DIM;
    __bf16* Qb  = (__bf16*)(ws + (size_t)9 * 1024 * 1024);
    __bf16* Kb  = Qb + (size_t)N_TOK * D_DIM;

    transpose_to_bf16<<<dim3(D_DIM / 32, N_TOK / 32), 256, 0, stream>>>(H, HbT, N_TOK, D_DIM);
    transpose_to_bf16<<<dim3(D_DIM / 32, D_DIM / 32), 256, 0, stream>>>(Wq, WqT, D_DIM, D_DIM);
    transpose_to_bf16<<<dim3(D_DIM / 32, D_DIM / 32), 256, 0, stream>>>(Wk, WkT, D_DIM, D_DIM);
    qk_gemm<<<1024, 256, 0, stream>>>(H, WqT, bq, WkT, bk, Qb, Kb);
    flash_attn<<<N_TOK / 32, 256, 0, stream>>>(H, Qb, Kb, HbT, out);
}
